// ExLRestSelfAtten_80547816669246
// MI455X (gfx1250) — matrix operand order repaired and checked by reading, not compiled, not run on hardware
//
#include <hip/hip_runtime.h>
#include <hip/hip_bf16.h>

typedef __attribute__((ext_vector_type(16))) _Float16 v16h;
typedef __attribute__((ext_vector_type(8)))  float    v8f;

#define SEQ      4096
#define ROWS_TOT 32768          // B*S = 8*4096
#define HID      128
#define TILE_R   64             // rows per block
#define STR      136            // LDS activation stride (f16), 272B rows -> 16B aligned
#define A_WIN    5
#define NW       11

union FragU {
    v16h v;
    _Float16 h[16];
    uint4 u4[2];
};

static __device__ __forceinline__ v8f wmma32(v16h a, v16h b, v8f c) {
    return __builtin_amdgcn_wmma_f32_16x16x32_f16(false, a, false, b, (short)0, c, false, false);
}

// ---- B-fragment layout in LDS --------------------------------------------
// For weight W[K][N] (row-major, row stride = N), fragment fs = nt*KS + kseg
// covers cols [nt*16, nt*16+16), K rows [kseg*32, kseg*32+32).
// lane L holds 16 f16: col n = nt*16 + (L&15), k = kseg*32 + (i>=8?16:0) + (L>=16?8:0) + (i&7).
static __device__ __forceinline__ void load_wfrags(_Float16* dst, const float* W,
                                                   int N, int NT, int KS,
                                                   int tid, int nthreads) {
    int total = NT * KS * 512;                 // 512 f16 per fragment
    for (int idx = tid; idx < total; idx += nthreads) {
        int i    = idx & 15;
        int lane = (idx >> 4) & 31;
        int fs   = idx >> 9;
        int nt   = fs / KS;
        int kseg = fs - nt * KS;
        int k = kseg * 32 + ((i >> 3) << 4) + ((lane >> 4) << 3) + (i & 7);
        int n = nt * 16 + (lane & 15);
        dst[idx] = (_Float16)W[k * N + n];
    }
}

static __device__ __forceinline__ v16h read_bfrag(const _Float16* wf, int fs, int lane) {
    FragU f;
    const _Float16* p = wf + ((fs * 32 + lane) << 4);
    f.u4[0] = *(const uint4*)p;
    f.u4[1] = *(const uint4*)(p + 8);
    return f.v;
}

// A fragment (16-bit A 16x32): lane<16 -> K = kk+{0..7} and kk+{16..23};
// lane>=16 -> +8. src rows at f16 stride 'str' (must make 16B alignment hold).
static __device__ __forceinline__ v16h read_afrag(const _Float16* src, int str,
                                                  int rowbase, int kk, int lane) {
    int g = lane >> 4;
    const _Float16* p = src + (rowbase + (lane & 15)) * str + kk + g * 8;
    FragU f;
    f.u4[0] = *(const uint4*)p;
    f.u4[1] = *(const uint4*)(p + 16);
    return f.v;
}

// One 16-row x (NT*16)-col GEMM layer out of LDS activations; epilogue writes
// f16 either back to LDS (stride dstStr) or to a global row-major [.,128] array.
template <int NT, int KS, bool TO_LDS>
static __device__ __forceinline__ void gemm_tile(const _Float16* src, int srcStr,
                                                 _Float16* dstL, int dstStr,
                                                 _Float16* dstG, size_t gRowBase,
                                                 const _Float16* wf, const float* bias,
                                                 bool relu, int lane, int wave) {
    v8f acc[NT] = {};
#pragma unroll
    for (int ks = 0; ks < KS; ++ks) {
        v16h a = read_afrag(src, srcStr, wave * 16, ks * 32, lane);
#pragma unroll
        for (int nt = 0; nt < NT; ++nt) {
            v16h b = read_bfrag(wf, nt * KS + ks, lane);
            acc[nt] = wmma32(a, b, acc[nt]);
        }
    }
    int g   = lane >> 4;
    int col = lane & 15;
#pragma unroll
    for (int nt = 0; nt < NT; ++nt) {
        float bv = bias ? bias[nt * 16 + col] : 0.0f;
#pragma unroll
        for (int i = 0; i < 8; ++i) {
            float val = acc[nt][i] + bv;
            if (relu) val = fmaxf(val, 0.0f);
            int row = wave * 16 + g * 8 + i;
            _Float16 hv = (_Float16)val;
            if (TO_LDS) dstL[row * dstStr + nt * 16 + col] = hv;
            else        dstG[(gRowBase + (size_t)row) * HID + nt * 16 + col] = hv;
        }
    }
}

// ---------------- Kernel 1: x -> h1 -> h2 -> q,k,v ------------------------
__global__ __launch_bounds__(128) void enc_kernel(
        const float* __restrict__ x,
        const float* __restrict__ m1, const float* __restrict__ b1,
        const float* __restrict__ m2, const float* __restrict__ b2,
        const float* __restrict__ wq, const float* __restrict__ wk,
        const float* __restrict__ wv,
        _Float16* __restrict__ qo, _Float16* __restrict__ ko,
        _Float16* __restrict__ vo) {
    __shared__ __align__(16) _Float16 xs[TILE_R * STR];       // 17408 B
    __shared__ __align__(16) _Float16 wf[8 * 4 * 512];        // 32768 B
    int tid  = threadIdx.x;
    int lane = tid & 31;
    int wave = tid >> 5;
    size_t r0 = (size_t)blockIdx.x * TILE_R;

    // load x tile (f32 -> f16)
    for (int idx = tid; idx < TILE_R * HID; idx += 128) {
        int row = idx >> 7, c = idx & 127;
        xs[row * STR + c] = (_Float16)x[(r0 + row) * HID + c];
    }
    load_wfrags(wf, m1, HID, 8, 4, tid, 128);
    __syncthreads();
    gemm_tile<8, 4, true>(xs, STR, xs, STR, nullptr, 0, wf, b1, true, lane, wave);
    __syncthreads();
    load_wfrags(wf, m2, HID, 8, 4, tid, 128);
    __syncthreads();
    gemm_tile<8, 4, true>(xs, STR, xs, STR, nullptr, 0, wf, b2, true, lane, wave);
    __syncthreads();
    load_wfrags(wf, wq, HID, 8, 4, tid, 128);
    __syncthreads();
    gemm_tile<8, 4, false>(xs, STR, nullptr, 0, qo, r0, wf, nullptr, false, lane, wave);
    __syncthreads();
    load_wfrags(wf, wk, HID, 8, 4, tid, 128);
    __syncthreads();
    gemm_tile<8, 4, false>(xs, STR, nullptr, 0, ko, r0, wf, nullptr, false, lane, wave);
    __syncthreads();
    load_wfrags(wf, wv, HID, 8, 4, tid, 128);
    __syncthreads();
    gemm_tile<8, 4, false>(xs, STR, nullptr, 0, vo, r0, wf, nullptr, false, lane, wave);
}

// ---------------- Kernel 2: windowed attention ----------------------------
// neighbor for window slot w (w=0..10) at seq pos s is s + 5 - w; out-of-range
// neighbors have zero key (=> dot 0 enters softmax) and zero value.
__global__ __launch_bounds__(128) void attn_kernel(
        const _Float16* __restrict__ q, const _Float16* __restrict__ k,
        const _Float16* __restrict__ v,
        float* __restrict__ out, _Float16* __restrict__ oatt) {
    __shared__ _Float16 qls[TILE_R * STR];          // 17408 B
    __shared__ _Float16 kls[(TILE_R + 16) * STR];   // 21760 B (74 rows used)
    __shared__ _Float16 vls[(TILE_R + 16) * STR];   // 21760 B
    int tid = threadIdx.x;
    size_t r0 = (size_t)blockIdx.x * TILE_R;
    size_t bbase = r0 & ~(size_t)(SEQ - 1);
    int s0 = (int)(r0 & (SEQ - 1));

    for (int idx = tid; idx < TILE_R * 64; idx += 128) {      // 2 f16 per u32
        int row = idx >> 6, cu = idx & 63;
        ((unsigned int*)&qls[row * STR])[cu] =
            ((const unsigned int*)&q[(r0 + row) * HID])[cu];
    }
    for (int idx = tid; idx < (TILE_R + 2 * A_WIN) * 64; idx += 128) {
        int row = idx >> 6, cu = idx & 63;
        int s = s0 - A_WIN + row;
        unsigned int kv = 0u, vv = 0u;
        if (s >= 0 && s < SEQ) {
            kv = ((const unsigned int*)&k[(bbase + (size_t)s) * HID])[cu];
            vv = ((const unsigned int*)&v[(bbase + (size_t)s) * HID])[cu];
        }
        ((unsigned int*)&kls[row * STR])[cu] = kv;
        ((unsigned int*)&vls[row * STR])[cu] = vv;
    }
    __syncthreads();

    int rl = tid >> 1, half = tid & 1;
    const _Float16* qp = &qls[rl * STR + half * 64];
    float d[NW];
#pragma unroll
    for (int w = 0; w < NW; ++w) {
        int n = rl + 2 * A_WIN - w;                 // halo row for neighbor s+5-w
        const _Float16* kp = &kls[n * STR + half * 64];
        float s = 0.0f;
        for (int c = 0; c < 64; ++c) s += (float)qp[c] * (float)kp[c];
        d[w] = s;
    }
    const float scale = 0.08838834764831845f;       // 1/sqrt(128)
#pragma unroll
    for (int w = 0; w < NW; ++w) d[w] = (d[w] + __shfl_xor(d[w], 1, 32)) * scale;

    float mx = d[0];
#pragma unroll
    for (int w = 1; w < NW; ++w) mx = fmaxf(mx, d[w]);
    float e[NW], sum = 0.0f;
#pragma unroll
    for (int w = 0; w < NW; ++w) { e[w] = __expf(d[w] - mx); sum += e[w]; }
    float inv = 1.0f / sum;
#pragma unroll
    for (int w = 0; w < NW; ++w) e[w] *= inv;

    if (half == 0) {
        float* ao = out + ROWS_TOT + (r0 + (size_t)rl) * NW;
#pragma unroll
        for (int w = 0; w < NW; ++w) ao[w] = e[w];
    }
    _Float16* op = oatt + (r0 + (size_t)rl) * HID + half * 64;
    for (int c = 0; c < 64; ++c) {
        float s = 0.0f;
#pragma unroll
        for (int w = 0; w < NW; ++w)
            s += e[w] * (float)vls[(rl + 2 * A_WIN - w) * STR + half * 64 + c];
        op[c] = (_Float16)s;
    }
}

// ---------------- Kernel 3: head MLP 128->64->32->1 -----------------------
__global__ __launch_bounds__(128) void head_kernel(
        const _Float16* __restrict__ oatt,
        const float* __restrict__ ml0, const float* __restrict__ bl0,
        const float* __restrict__ ml1, const float* __restrict__ bl1,
        const float* __restrict__ m3, const float* __restrict__ b3,
        float* __restrict__ out) {
    __shared__ __align__(16) _Float16 os[TILE_R * STR];   // 17408 B
    __shared__ __align__(16) _Float16 wf0[4 * 4 * 512];   // 16384 B
    __shared__ __align__(16) _Float16 h3[TILE_R * 72];    //  9216 B
    __shared__ __align__(16) _Float16 wf1[2 * 2 * 512];   //  4096 B
    __shared__ __align__(16) _Float16 h4[TILE_R * 40];    //  5120 B
    int tid  = threadIdx.x;
    int lane = tid & 31;
    int wave = tid >> 5;
    size_t r0 = (size_t)blockIdx.x * TILE_R;

    for (int idx = tid; idx < TILE_R * 64; idx += 128) {
        int row = idx >> 6, cu = idx & 63;
        ((unsigned int*)&os[row * STR])[cu] =
            ((const unsigned int*)&oatt[(r0 + row) * HID])[cu];
    }
    load_wfrags(wf0, ml0, 64, 4, 4, tid, 128);
    __syncthreads();
    gemm_tile<4, 4, true>(os, STR, h3, 72, nullptr, 0, wf0, bl0, true, lane, wave);
    __syncthreads();
    load_wfrags(wf1, ml1, 32, 2, 2, tid, 128);
    __syncthreads();
    gemm_tile<2, 2, true>(h3, 72, h4, 40, nullptr, 0, wf1, bl1, true, lane, wave);
    __syncthreads();
    if (tid < TILE_R) {
        float s = b3[0];
        for (int c = 0; c < 32; ++c) s += (float)h4[tid * 40 + c] * m3[c];
        out[r0 + (size_t)tid] = 1.0f / (1.0f + __expf(-s));
    }
}

extern "C" void kernel_launch(void* const* d_in, const int* in_sizes, int n_in,
                              void* d_out, int out_size, void* d_ws, size_t ws_size,
                              hipStream_t stream) {
    const float* x   = (const float*)d_in[0];
    const float* m1  = (const float*)d_in[1];
    const float* b1  = (const float*)d_in[2];
    const float* m2  = (const float*)d_in[3];
    const float* b2  = (const float*)d_in[4];
    const float* wq  = (const float*)d_in[5];
    const float* wk  = (const float*)d_in[6];
    const float* wv  = (const float*)d_in[7];
    const float* ml0 = (const float*)d_in[8];
    const float* bl0 = (const float*)d_in[9];
    const float* ml1 = (const float*)d_in[10];
    const float* bl1 = (const float*)d_in[11];
    const float* m3  = (const float*)d_in[12];
    const float* b3  = (const float*)d_in[13];
    float* out = (float*)d_out;

    _Float16* q    = (_Float16*)d_ws;
    _Float16* k    = q + (size_t)ROWS_TOT * HID;
    _Float16* v    = k + (size_t)ROWS_TOT * HID;
    _Float16* oatt = v + (size_t)ROWS_TOT * HID;

    dim3 grid(ROWS_TOT / TILE_R), block(128);
    enc_kernel<<<grid, block, 0, stream>>>(x, m1, b1, m2, b2, wq, wk, wv, q, k, v);
    attn_kernel<<<grid, block, 0, stream>>>(q, k, v, out, oatt);
    head_kernel<<<grid, block, 0, stream>>>(oatt, ml0, bl0, ml1, bl1, m3, b3, out);
}
